// LongformerHuggingface_64304250355909
// MI455X (gfx1250) — matrix operand order repaired and checked by reading, not compiled, not run on hardware
//
#include <hip/hip_runtime.h>
#include <cstdint>

typedef __bf16 bf16_t;
typedef __attribute__((ext_vector_type(16))) __bf16 v16bf;
typedef __attribute__((ext_vector_type(8)))  __bf16 v8bf;
typedef __attribute__((ext_vector_type(8)))  float  v8f;

#define BB   2
#define SS   4096
#define DD   768
#define HH   12
#define DHH  64
#define W1W  256
#define NCC  16

// Low 32 bits of a generic pointer to LDS == LDS byte address (flat aperture rule).
static __device__ __forceinline__ uint32_t lds_addr(const void* p) {
  return (uint32_t)(uintptr_t)p;
}

// CDNA5 async global->LDS copy, 16 bytes per lane, GVS mode (saddr + 32-bit voffset).
// Tracked by ASYNCcnt; in-order per wave, so partial-count waits are exact.
static __device__ __forceinline__ void async_copy16(uint32_t lds, uint32_t goff, uint64_t sbase) {
  asm volatile("global_load_async_to_lds_b128 %0, %1, %2"
               :: "v"(lds), "v"(goff), "s"(sbase) : "memory");
}
static __device__ __forceinline__ void wait_async0() {
  asm volatile("s_wait_asynccnt 0" ::: "memory");
}
static __device__ __forceinline__ void wait_async2() {   // <=2 outstanding
  asm volatile("s_wait_asynccnt 2" ::: "memory");
}
static __device__ __forceinline__ void wait_async4() {   // <=4 outstanding
  asm volatile("s_wait_asynccnt 4" ::: "memory");
}

// Build a v16bf WMMA fragment from two contiguous 16-byte chunks.
// A-layout: chunks at (16*half) and (16*half + 8) -> K = j + 16*half (same as B)
// B-layout: chunks at (16*half) and (16*half + 8) -> K = j + 16*half
static __device__ __forceinline__ v16bf frag_from_chunks(const bf16_t* p0, const bf16_t* p1) {
  v8bf lo = *(const v8bf*)p0;
  v8bf hi = *(const v8bf*)p1;
  v16bf r;
#pragma unroll
  for (int i = 0; i < 8; ++i) { r[i] = lo[i]; r[i + 8] = hi[i]; }
  return r;
}

// ---------------- embeddings + LayerNorm ----------------
__global__ __launch_bounds__(256) void embed_ln_kernel(
    const int* __restrict__ ids, const float* __restrict__ we,
    const float* __restrict__ pe, const float* __restrict__ g,
    const float* __restrict__ be, float* __restrict__ x, bf16_t* __restrict__ xh)
{
  int t = blockIdx.x;                 // token index 0..B*S-1
  int s = t & (SS - 1);
  int id = ids[t];
  __shared__ float sh[16];
  float v[3]; float sum = 0.f, sq = 0.f;
#pragma unroll
  for (int i = 0; i < 3; ++i) {
    int idx = threadIdx.x + i * 256;
    float val = we[(size_t)id * DD + idx] + pe[(size_t)s * DD + idx];
    v[i] = val; sum += val; sq += val * val;
  }
#pragma unroll
  for (int off = 16; off > 0; off >>= 1) {
    sum += __shfl_xor(sum, off, 32);
    sq  += __shfl_xor(sq,  off, 32);
  }
  int wv = threadIdx.x >> 5;
  if ((threadIdx.x & 31) == 0) { sh[wv] = sum; sh[wv + 8] = sq; }
  __syncthreads();
  float ts = 0.f, tq = 0.f;
#pragma unroll
  for (int i = 0; i < 8; ++i) { ts += sh[i]; tq += sh[i + 8]; }
  float mean = ts / (float)DD;
  float var  = tq / (float)DD - mean * mean;
  float rstd = rsqrtf(var + 1e-5f);
#pragma unroll
  for (int i = 0; i < 3; ++i) {
    int idx = threadIdx.x + i * 256;
    float y = (v[i] - mean) * rstd * g[idx] + be[idx];
    x [(size_t)t * DD + idx] = y;
    xh[(size_t)t * DD + idx] = (bf16_t)y;
  }
}

// ---------------- final LayerNorm ----------------
__global__ __launch_bounds__(256) void ln_out_kernel(
    const float* __restrict__ hin, const float* __restrict__ g,
    const float* __restrict__ be, float* __restrict__ out)
{
  int t = blockIdx.x;
  __shared__ float sh[16];
  float v[3]; float sum = 0.f, sq = 0.f;
#pragma unroll
  for (int i = 0; i < 3; ++i) {
    int idx = threadIdx.x + i * 256;
    float val = hin[(size_t)t * DD + idx];
    v[i] = val; sum += val; sq += val * val;
  }
#pragma unroll
  for (int off = 16; off > 0; off >>= 1) {
    sum += __shfl_xor(sum, off, 32);
    sq  += __shfl_xor(sq,  off, 32);
  }
  int wv = threadIdx.x >> 5;
  if ((threadIdx.x & 31) == 0) { sh[wv] = sum; sh[wv + 8] = sq; }
  __syncthreads();
  float ts = 0.f, tq = 0.f;
#pragma unroll
  for (int i = 0; i < 8; ++i) { ts += sh[i]; tq += sh[i + 8]; }
  float mean = ts / (float)DD;
  float var  = tq / (float)DD - mean * mean;
  float rstd = rsqrtf(var + 1e-5f);
#pragma unroll
  for (int i = 0; i < 3; ++i) {
    int idx = threadIdx.x + i * 256;
    out[(size_t)t * DD + idx] = (v[i] - mean) * rstd * g[idx] + be[idx];
  }
}

// ---------------- fp32 -> bf16 weight conversion WITH transpose ----------------
__global__ __launch_bounds__(256) void cvt_w_kernel(
    const float* __restrict__ a, const float* __restrict__ b,
    const float* __restrict__ c, const float* __restrict__ d,
    bf16_t* __restrict__ oa, bf16_t* __restrict__ ob,
    bf16_t* __restrict__ oc, bf16_t* __restrict__ od)
{
  int i = blockIdx.x * 256 + threadIdx.x;
  if (i >= DD * DD) return;
  int k = i / DD, n = i - k * DD;     // source [k][n] (coalesced read)
  size_t o = (size_t)n * DD + k;      // dest   [n][k]
  oa[o] = (bf16_t)a[i]; ob[o] = (bf16_t)b[i];
  oc[o] = (bf16_t)c[i]; od[o] = (bf16_t)d[i];
}

// ---------------- WMMA GEMM (double-buffered async pipeline) ----------------
// Wt is pre-transposed [N][K].
// MODE 0: out_h[b,h,s,dh]  = y*scale (bf16, head layout)     -- q, k
// MODE 2: out_h[b,h,dh,s]  = y       (bf16, transposed head) -- v
// MODE 1: out_f[row,col]   = y + residual[row,col] (fp32)    -- O-proj
template <int MODE>
__global__ __launch_bounds__(128) void gemm_wmma_kernel(
    const bf16_t* __restrict__ A, const bf16_t* __restrict__ Wt,
    const float* __restrict__ bias, const float* __restrict__ residual,
    bf16_t* __restrict__ out_h, float* __restrict__ out_f, float scale)
{
  __shared__ bf16_t As[2][64][32];   // [buf][m][k]
  __shared__ bf16_t Bs[2][64][32];   // [buf][n][k]
  const int tid  = threadIdx.x;
  const int lane = tid & 31, wv = tid >> 5;
  const int half = lane >> 4, l16 = lane & 15;
  const int m0 = blockIdx.y * 64, n0 = blockIdx.x * 64;
  const int wm = (wv >> 1) * 32,  wn = (wv & 1) * 32;

  v8f acc[2][2] = {};
  const int srow = tid >> 1, scg = (tid & 1) * 16;   // staging: 64 rows x 32 cols
  const uint64_t abase = (uint64_t)A;
  const uint64_t wbase = (uint64_t)Wt;

  auto stage = [&](int k0, int buf) {
    uint32_t ao = (uint32_t)(((m0 + srow) * DD + k0 + scg) * 2);
    async_copy16(lds_addr(&As[buf][srow][scg]),     ao,      abase);
    async_copy16(lds_addr(&As[buf][srow][scg + 8]), ao + 16, abase);
    uint32_t bo = (uint32_t)(((n0 + srow) * DD + k0 + scg) * 2);
    async_copy16(lds_addr(&Bs[buf][srow][scg]),     bo,      wbase);
    async_copy16(lds_addr(&Bs[buf][srow][scg + 8]), bo + 16, wbase);
  };

  const int NIT = DD / 32;  // 24
  stage(0, 0);
  for (int i = 0; i < NIT; ++i) {
    if (i + 1 < NIT) { stage((i + 1) * 32, (i + 1) & 1); wait_async4(); }
    else             { wait_async0(); }
    __syncthreads();
    const int buf = i & 1;

    v16bf af[2], bfr[2];
#pragma unroll
    for (int mt = 0; mt < 2; ++mt) {
      const bf16_t* r = &As[buf][wm + mt * 16 + l16][0];
      af[mt] = frag_from_chunks(r + 16 * half, r + 16 * half + 8);
    }
#pragma unroll
    for (int nt = 0; nt < 2; ++nt) {
      const bf16_t* r = &Bs[buf][wn + nt * 16 + l16][0];
      bfr[nt] = frag_from_chunks(r + 16 * half, r + 16 * half + 8);
    }
#pragma unroll
    for (int mt = 0; mt < 2; ++mt)
#pragma unroll
      for (int nt = 0; nt < 2; ++nt)
        acc[mt][nt] = __builtin_amdgcn_wmma_f32_16x16x32_bf16(
            false, af[mt], false, bfr[nt], (short)0, acc[mt][nt], false, false);
    __syncthreads();   // all waves done reading buf before it is overwritten
  }

#pragma unroll
  for (int mt = 0; mt < 2; ++mt) {
#pragma unroll
    for (int nt = 0; nt < 2; ++nt) {
      int ncol = n0 + wn + nt * 16 + l16;
      float bb = bias[ncol];
#pragma unroll
      for (int r = 0; r < 8; ++r) {
        int row = m0 + wm + mt * 16 + r + 8 * half;
        float val = acc[mt][nt][r] + bb;
        if constexpr (MODE == 0) {
          val *= scale;
          int b = row >> 12, s = row & (SS - 1);
          int h = ncol >> 6, dh = ncol & 63;
          out_h[(((size_t)b * HH + h) * SS + s) * DHH + dh] = (bf16_t)val;
        } else if constexpr (MODE == 2) {
          int b = row >> 12, s = row & (SS - 1);
          int h = ncol >> 6, dh = ncol & 63;
          out_h[(((size_t)b * HH + h) * DHH + dh) * SS + s] = (bf16_t)val;
        } else {
          val += residual[(size_t)row * DD + ncol];
          out_f[(size_t)row * DD + ncol] = val;
        }
      }
    }
  }
}

// ---------------- banded flash attention (double-buffered async pipeline) ----------------
// grid: (B*H, NC, 2). Block = 256 thr = 8 waves; each wave owns 16 query rows.
// kh: [B,H,S,DH]; vt: [B,H,DH,S] (pre-transposed by the V projection epilogue).
// A block's 128 queries only see 640 band keys -> exactly 20 key tiles starting
// at tile qb*4 (instead of all 24).
__global__ __launch_bounds__(256) void attn_wmma_kernel(
    const bf16_t* __restrict__ qh, const bf16_t* __restrict__ kh,
    const bf16_t* __restrict__ vt, bf16_t* __restrict__ ctx)
{
  __shared__ bf16_t Ks[2][32][64];    // [buf][key][d]
  __shared__ bf16_t Vt[2][64][32];    // [buf][d][key]
  __shared__ bf16_t Ps[8][16][32];    // per-wave probs [row][key]

  const int tid  = threadIdx.x;
  const int lane = tid & 31, wid = tid >> 5;
  const int half = lane >> 4, l16 = lane & 15;
  const int bh = blockIdx.x;          // b*H + h
  const int c  = blockIdx.y;          // chunk
  const int qb = blockIdx.z;          // query half of chunk
  const int b  = bh / HH, h = bh % HH;
  const int kg0   = (c - 1) * W1W;            // global pos of band key 0 (may be <0)
  const int qloc0 = qb * 128 + wid * 16;      // local query base for this wave
  const int sq    = c * W1W + qloc0 + l16;    // this lane's query row (A-frag)

  const bf16_t* qrow = qh + ((size_t)bh * SS + sq) * DHH;
  v16bf qa[2];
  qa[0] = frag_from_chunks(qrow + 16 * half,      qrow + 16 * half + 8);
  qa[1] = frag_from_chunks(qrow + 32 + 16 * half, qrow + 32 + 16 * half + 8);

  float mr[8], lr[8];
#pragma unroll
  for (int r = 0; r < 8; ++r) { mr[r] = -3.0e38f; lr[r] = 0.f; }
  v8f oacc[4] = {};

  const int skey = tid >> 3, sdg = (tid & 7) * 8;   // K staging: key 0..31, d group
  const int svd  = tid >> 2, svk = (tid & 3) * 8;   // V staging: d 0..63, key group
  const uint64_t kbase = (uint64_t)kh;
  const uint64_t vbase = (uint64_t)vt;

  // Out-of-band rows are CLAMPED (band mask kills them later): no divergence.
  auto stage = [&](int kt, int buf) {
    int kl0 = kt * 32;
    int sg  = kg0 + kl0 + skey;
    int sgc = min(max(sg, 0), SS - 1);
    uint32_t ko = (uint32_t)((((bh * SS + sgc) * DHH) + sdg) * 2);
    async_copy16(lds_addr(&Ks[buf][skey][sdg]), ko, kbase);
    int vs  = kg0 + kl0 + svk;
    int vsc = min(max(vs, 0), SS - 8);
    uint32_t vo = (uint32_t)((((bh * DHH + svd) * SS) + vsc) * 2);
    async_copy16(lds_addr(&Vt[buf][svd][svk]), vo, vbase);
  };

  const int kt0 = qb * 4;   // first key tile touching this block's queries
  const int NT  = 20;       // 640 band keys / 32
  stage(kt0, 0);
  for (int i = 0; i < NT; ++i) {
    if (i + 1 < NT) { stage(kt0 + i + 1, (i + 1) & 1); wait_async2(); }
    else            { wait_async0(); }
    __syncthreads();
    const int buf = i & 1;
    const int kl0 = (kt0 + i) * 32;

    // scores: S = Q(16x64) @ K^T, two 16-key halves, K-dim 64 = 2 chained WMMAs
    v8f s0 = {}, s1 = {};
    {
      const bf16_t* r0 = &Ks[buf][l16][0];
      const bf16_t* r1 = &Ks[buf][16 + l16][0];
      v16bf kb;
      kb = frag_from_chunks(r0 + 16 * half, r0 + 16 * half + 8);
      s0 = __builtin_amdgcn_wmma_f32_16x16x32_bf16(false, qa[0], false, kb, (short)0, s0, false, false);
      kb = frag_from_chunks(r0 + 32 + 16 * half, r0 + 32 + 16 * half + 8);
      s0 = __builtin_amdgcn_wmma_f32_16x16x32_bf16(false, qa[1], false, kb, (short)0, s0, false, false);
      kb = frag_from_chunks(r1 + 16 * half, r1 + 16 * half + 8);
      s1 = __builtin_amdgcn_wmma_f32_16x16x32_bf16(false, qa[0], false, kb, (short)0, s1, false, false);
      kb = frag_from_chunks(r1 + 32 + 16 * half, r1 + 32 + 16 * half + 8);
      s1 = __builtin_amdgcn_wmma_f32_16x16x32_bf16(false, qa[1], false, kb, (short)0, s1, false, false);
    }

    // band mask + online softmax (C-layout: row = r + 8*half, col = l16)
    const int k0l = kl0 + l16, k1l = k0l + 16;
    const bool kv0 = (kg0 + k0l >= 0) && (kg0 + k0l < SS);   // tile-invariant part
    const bool kv1 = (kg0 + k1l >= 0) && (kg0 + k1l < SS);
#pragma unroll
    for (int r = 0; r < 8; ++r) {
      int ql = qloc0 + r + 8 * half;
      bool v0 = kv0 && (k0l >= ql) && (k0l <= ql + 2 * W1W);
      bool v1 = kv1 && (k1l >= ql) && (k1l <= ql + 2 * W1W);
      float a0 = v0 ? s0[r] : -3.0e38f;
      float a1 = v1 ? s1[r] : -3.0e38f;
      float t = fmaxf(a0, a1);
#pragma unroll
      for (int off = 8; off > 0; off >>= 1) t = fmaxf(t, __shfl_xor(t, off, 16));
      float mnew  = fmaxf(mr[r], t);
      float alpha = __expf(mr[r] - mnew);
      float p0 = v0 ? __expf(a0 - mnew) : 0.f;
      float p1 = v1 ? __expf(a1 - mnew) : 0.f;
      float rs = p0 + p1;
#pragma unroll
      for (int off = 8; off > 0; off >>= 1) rs += __shfl_xor(rs, off, 16);
      lr[r] = lr[r] * alpha + rs;
      mr[r] = mnew;
#pragma unroll
      for (int nt = 0; nt < 4; ++nt) oacc[nt][r] *= alpha;
      Ps[wid][r + 8 * half][l16]      = (bf16_t)p0;
      Ps[wid][r + 8 * half][l16 + 16] = (bf16_t)p1;
    }

    // O += P(16x32) @ V(32x64): A-frag from per-wave LDS, 4 WMMAs over d tiles
    {
      const bf16_t* pr = &Ps[wid][l16][0];
      v16bf pa = frag_from_chunks(pr + 16 * half, pr + 16 * half + 8);
#pragma unroll
      for (int nt = 0; nt < 4; ++nt) {
        const bf16_t* vrow = &Vt[buf][nt * 16 + l16][0];
        v16bf vb = frag_from_chunks(vrow + 16 * half, vrow + 16 * half + 8);
        oacc[nt] = __builtin_amdgcn_wmma_f32_16x16x32_bf16(
            false, pa, false, vb, (short)0, oacc[nt], false, false);
      }
    }
    __syncthreads();   // all waves done reading buf before it is overwritten
  }

  // normalize and write ctx [B,S,D]
#pragma unroll
  for (int nt = 0; nt < 4; ++nt) {
    int d = nt * 16 + l16;
#pragma unroll
    for (int r = 0; r < 8; ++r) {
      int sidx = c * W1W + qloc0 + r + 8 * half;
      float val = oacc[nt][r] / lr[r];
      ctx[((size_t)b * SS + sidx) * DD + h * DHH + d] = (bf16_t)val;
    }
  }
}

extern "C" void kernel_launch(void* const* d_in, const int* in_sizes, int n_in,
                              void* d_out, int out_size, void* d_ws, size_t ws_size,
                              hipStream_t stream) {
  const int*   ids   = (const int*)d_in[0];
  const float* state = (const float*)d_in[1];
  const float* we    = (const float*)d_in[2];
  const float* pe    = (const float*)d_in[3];
  const float* lneg  = (const float*)d_in[4];
  const float* lneb  = (const float*)d_in[5];
  const float* Wq    = (const float*)d_in[6];
  const float* bq    = (const float*)d_in[7];
  const float* Wk    = (const float*)d_in[8];
  const float* bk    = (const float*)d_in[9];
  const float* Wv    = (const float*)d_in[10];
  const float* bv    = (const float*)d_in[11];
  const float* Wo    = (const float*)d_in[12];
  const float* bo    = (const float*)d_in[13];
  const float* lnag  = (const float*)d_in[14];
  const float* lnab  = (const float*)d_in[15];

  char* ws = (char*)d_ws;
  size_t off = 0;
  auto alloc = [&](size_t bytes) -> char* {
    char* p = ws + off;
    off += (bytes + 255) & ~(size_t)255;
    return p;
  };
  const size_t MT = (size_t)BB * SS;   // 8192 tokens
  float*  x_f  = (float*)alloc(MT * DD * sizeof(float));
  float*  hid  = (float*)alloc(MT * DD * sizeof(float));
  bf16_t* x_h  = (bf16_t*)alloc(MT * DD * 2);
  bf16_t* q_h  = (bf16_t*)alloc(MT * DD * 2);
  bf16_t* k_h  = (bf16_t*)alloc(MT * DD * 2);
  bf16_t* v_t  = (bf16_t*)alloc(MT * DD * 2);
  bf16_t* c_h  = (bf16_t*)alloc(MT * DD * 2);
  bf16_t* wqt  = (bf16_t*)alloc((size_t)DD * DD * 2);
  bf16_t* wkt  = (bf16_t*)alloc((size_t)DD * DD * 2);
  bf16_t* wvt  = (bf16_t*)alloc((size_t)DD * DD * 2);
  bf16_t* wot  = (bf16_t*)alloc((size_t)DD * DD * 2);

  cvt_w_kernel<<<(DD * DD + 255) / 256, 256, 0, stream>>>(Wq, Wk, Wv, Wo,
                                                          wqt, wkt, wvt, wot);
  embed_ln_kernel<<<(int)MT, 256, 0, stream>>>(ids, we, pe, lneg, lneb, x_f, x_h);

  dim3 gg(DD / 64, (unsigned)(MT / 64));
  gemm_wmma_kernel<0><<<gg, 128, 0, stream>>>(x_h, wqt, bq, nullptr, q_h, nullptr, 0.125f);
  gemm_wmma_kernel<0><<<gg, 128, 0, stream>>>(x_h, wkt, bk, nullptr, k_h, nullptr, 1.0f);
  gemm_wmma_kernel<2><<<gg, 128, 0, stream>>>(x_h, wvt, bv, nullptr, v_t, nullptr, 1.0f);

  attn_wmma_kernel<<<dim3(BB * HH, NCC, 2), 256, 0, stream>>>(q_h, k_h, v_t, c_h);

  gemm_wmma_kernel<1><<<gg, 128, 0, stream>>>(c_h, wot, bo, x_f, nullptr, hid, 1.0f);
  ln_out_kernel<<<(int)MT, 256, 0, stream>>>(hid, lnag, lnab, (float*)d_out);

  hipMemcpyAsync((float*)d_out + MT * DD, state, (size_t)in_sizes[1] * sizeof(float),
                 hipMemcpyDeviceToDevice, stream);
}
